// PointNet2_58102317580438
// MI455X (gfx1250) — compile-verified, matrix operand order checked by reading, not run on hardware
//
#include <hip/hip_runtime.h>

typedef __attribute__((ext_vector_type(16))) _Float16 v16h;
typedef __attribute__((ext_vector_type(8)))  float    v8f;

#define BB   4
#define NPTS 8192
#define KNB  32
#define MS1  4096
#define MS2  1024
#define NCLS 40
#define BIGF 1e10f

// ---------------------------------------------------------------------------
// Farthest point sampling: one block per batch, dist kept in registers.
// ---------------------------------------------------------------------------
template<int NSRC, int PPT>
__global__ __launch_bounds__(NSRC / PPT)
void fps_kernel(const float* __restrict__ pos, int* __restrict__ outIdx, int m_samples)
{
    constexpr int T = NSRC / PPT;
    const int b = blockIdx.x;
    const int tid = threadIdx.x;
    const float* P = pos + (size_t)b * NSRC * 3;

    float px[PPT], py[PPT], pz[PPT], dist[PPT];
    #pragma unroll
    for (int i = 0; i < PPT; ++i) {
        int p = tid + i * T;
        px[i] = P[p * 3 + 0]; py[i] = P[p * 3 + 1]; pz[i] = P[p * 3 + 2];
        dist[i] = BIGF;
    }

    __shared__ float s_val[32];
    __shared__ int   s_idx[32];
    __shared__ int   s_last;
    __shared__ float s_lp[3];
    if (tid == 0) s_last = 0;
    __syncthreads();

    for (int m = 0; m < m_samples; ++m) {
        int last = s_last;
        if (tid == 0) {
            outIdx[(size_t)b * m_samples + m] = last;
            s_lp[0] = P[last * 3 + 0];
            s_lp[1] = P[last * 3 + 1];
            s_lp[2] = P[last * 3 + 2];
        }
        __syncthreads();
        float lx = s_lp[0], ly = s_lp[1], lz = s_lp[2];
        float bestV = -1.0f; int bestI = 0x7fffffff;
        #pragma unroll
        for (int i = 0; i < PPT; ++i) {
            float dx = px[i] - lx, dy = py[i] - ly, dz = pz[i] - lz;
            float d = dx * dx + dy * dy + dz * dz;
            dist[i] = fminf(dist[i], d);
            int p = tid + i * T;
            if (dist[i] > bestV || (dist[i] == bestV && p < bestI)) { bestV = dist[i]; bestI = p; }
        }
        #pragma unroll
        for (int off = 16; off >= 1; off >>= 1) {
            float ov = __shfl_xor(bestV, off, 32);
            int   oi = __shfl_xor(bestI, off, 32);
            if (ov > bestV || (ov == bestV && oi < bestI)) { bestV = ov; bestI = oi; }
        }
        if ((tid & 31) == 0) { s_val[tid >> 5] = bestV; s_idx[tid >> 5] = bestI; }
        __syncthreads();
        if (tid < 32) {
            int nw = T >> 5;
            float v = (tid < nw) ? s_val[tid] : -1.0f;
            int   i2 = (tid < nw) ? s_idx[tid] : 0x7fffffff;
            #pragma unroll
            for (int off = 16; off >= 1; off >>= 1) {
                float ov = __shfl_xor(v, off, 32);
                int   oi = __shfl_xor(i2, off, 32);
                if (ov > v || (ov == v && oi < i2)) { v = ov; i2 = oi; }
            }
            if (tid == 0) s_last = i2;
        }
        __syncthreads();
    }
}

// ---------------------------------------------------------------------------
// Gather sampled centers.
// ---------------------------------------------------------------------------
__global__ void gather_centers_kernel(const float* __restrict__ src, const int* __restrict__ idx,
                                      float* __restrict__ dst, int M, int Nsrc)
{
    int t = blockIdx.x * blockDim.x + threadIdx.x;
    if (t >= BB * M) return;
    int b = t / M, m = t % M;
    int j = idx[(size_t)b * M + m];
    const float* s = src + ((size_t)b * Nsrc + j) * 3;
    float* d = dst + ((size_t)b * M + m) * 3;
    d[0] = s[0]; d[1] = s[1]; d[2] = s[2];
}

// ---------------------------------------------------------------------------
// Ball query: one wave per center; ballot-compaction of in-radius neighbors
// (index order, up to K=32), validity returned as a 32-bit lane mask.
// ---------------------------------------------------------------------------
__global__ __launch_bounds__(128)
void ball_query_kernel(const float* __restrict__ src, const float* __restrict__ ctr,
                       int* __restrict__ nidx, unsigned* __restrict__ vmask,
                       int M, int Nsrc, float r2)
{
    const int wavesPerBlk = blockDim.x >> 5;
    const int gw = blockIdx.x * wavesPerBlk + (threadIdx.x >> 5);
    const int lane = threadIdx.x & 31;
    if (gw >= BB * M) return;
    const int b = gw / M, m = gw % M;
    const float* S = src + (size_t)b * Nsrc * 3;
    const float* C = ctr + ((size_t)b * M + m) * 3;
    const float cx = C[0], cy = C[1], cz = C[2];
    int* out = nidx + (size_t)gw * KNB;

    int found = 0;
    for (int base = 0; base < Nsrc && found < KNB; base += 32) {
        int j = base + lane;
        float dx = S[j * 3 + 0] - cx, dy = S[j * 3 + 1] - cy, dz = S[j * 3 + 2] - cz;
        float d2 = dx * dx + dy * dy + dz * dz;
        bool pred = d2 <= r2;
        unsigned mask = (unsigned)__ballot(pred);
        int before = __popc(mask & ((1u << lane) - 1u));
        int slot = found + before;
        if (pred && slot < KNB) out[slot] = j;
        found += __popc(mask);
    }
    if (found > KNB) found = KNB;
    if (lane >= found) out[lane] = 0;
    if (lane == 0)
        vmask[gw] = (found >= 32) ? 0xFFFFFFFFu : ((1u << found) - 1u);
}

// ---------------------------------------------------------------------------
// Repack weights [Cin][Cout] f32 -> fragment-major f16 B-fragments.
// Fragment (kk,nn) holds 32 lanes x 16 halves contiguous:
//   K = kk*32 + (lane<16 ? 0 : 8) + (e<8 ? e : e+8),  N = nn*16 + (lane&15)
// ---------------------------------------------------------------------------
__global__ void repack_w_kernel(const float* __restrict__ W, _Float16* __restrict__ F,
                                int Cin, int Cout, int CinPad)
{
    int t = blockIdx.x * blockDim.x + threadIdx.x;
    int total = (CinPad / 32) * (Cout / 16) * 512;
    if (t >= total) return;
    int e = t & 15;
    int lane = (t >> 4) & 31;
    int f = t >> 9;
    int nTiles = Cout / 16;
    int kk = f / nTiles, nn = f % nTiles;
    int K = kk * 32 + ((lane < 16) ? 0 : 8) + ((e < 8) ? e : e + 8);
    int N = nn * 16 + (lane & 15);
    float v = (K < Cin) ? W[(size_t)K * Cout + N] : 0.0f;
    F[t] = (_Float16)v;
}

// Fold BN-eval affine: relu((z+b)*s+t) == relu(z*scale + shift)
__global__ void fold_affine_kernel(const float* __restrict__ bb, const float* __restrict__ ss,
                                   const float* __restrict__ tt,
                                   float* __restrict__ scale, float* __restrict__ shift, int C)
{
    int i = blockIdx.x * blockDim.x + threadIdx.x;
    if (i >= C) return;
    scale[i] = ss[i];
    shift[i] = bb[i] * ss[i] + tt[i];
}

__global__ void zero_kernel(float* __restrict__ p, int n)
{
    int t = blockIdx.x * blockDim.x + threadIdx.x;
    if (t < n) p[t] = 0.0f;
}

// ---------------------------------------------------------------------------
// Fused PointConv: one wave per center. 32 neighbors = 2 WMMA row tiles.
// gather->f16 LDS -> WMMA MLP layer1 -> LDS -> WMMA MLP layer2 -> masked max.
// ---------------------------------------------------------------------------
template<int CINP, int C1, int C2, int XD>
__global__ __launch_bounds__(32)
void pointconv_kernel(const float* __restrict__ xfeat,   // [B][Nsrc][XD]
                      const float* __restrict__ srcpos,  // [B][Nsrc][3]
                      const float* __restrict__ ctr,     // [B][M][3]
                      const int* __restrict__ nidx,      // [B][M][K]
                      const unsigned* __restrict__ vmask,
                      const _Float16* __restrict__ F1,
                      const float* __restrict__ sc1, const float* __restrict__ sh1,
                      const _Float16* __restrict__ F2,
                      const float* __restrict__ sc2, const float* __restrict__ sh2,
                      float* __restrict__ xout,          // [B][M][C2]
                      int M, int Nsrc)
{
    constexpr int KS1 = CINP / 32;
    constexpr int KS2 = C1 / 32;
    __shared__ _Float16 Araw[32][CINP];
    __shared__ _Float16 H1[32][C1];

    const int gw = blockIdx.x;
    const int lane = threadIdx.x;
    const int b = gw / M;

    const int* idx = nidx + (size_t)gw * KNB;
    const unsigned valid = vmask[gw];
    const float* C = ctr + (size_t)gw * 3;
    const float cx = C[0], cy = C[1], cz = C[2];

    // gather one neighbor row per lane
    {
        int j = idx[lane];
        const float* xv = xfeat + ((size_t)b * Nsrc + j) * XD;
        for (int c = 0; c < XD; ++c)
            Araw[lane][c] = (_Float16)xv[c];
        const float* pv = srcpos + ((size_t)b * Nsrc + j) * 3;
        Araw[lane][XD + 0] = (_Float16)(pv[0] - cx);
        Araw[lane][XD + 1] = (_Float16)(pv[1] - cy);
        Araw[lane][XD + 2] = (_Float16)(pv[2] - cz);
        for (int c = XD + 3; c < CINP; ++c)
            Araw[lane][c] = (_Float16)0.0f;
    }
    __syncthreads();

    const int arow  = lane & 15;
    const int kbase = (lane < 16) ? 0 : 8;
    const int col   = lane & 15;
    const int rowhi = (lane < 16) ? 0 : 8;

    // ---- layer 1: A fragments (2 row tiles x KS1) ----
    v16h a0[2][KS1];
    #pragma unroll
    for (int t = 0; t < 2; ++t)
        #pragma unroll
        for (int kk = 0; kk < KS1; ++kk) {
            v16h a;
            #pragma unroll
            for (int e = 0; e < 16; ++e) {
                int K = kk * 32 + kbase + ((e < 8) ? e : e + 8);
                a[e] = Araw[t * 16 + arow][K];
            }
            a0[t][kk] = a;
        }

    #pragma unroll 1
    for (int nn = 0; nn < C1 / 16; ++nn) {
        float scale = sc1[nn * 16 + col];
        float shift = sh1[nn * 16 + col];
        #pragma unroll
        for (int t = 0; t < 2; ++t) {
            v8f acc = {};
            #pragma unroll
            for (int kk = 0; kk < KS1; ++kk) {
                v16h bfrag = *(const v16h*)(F1 + ((size_t)(kk * (C1 / 16) + nn)) * 512 + lane * 16);
                acc = __builtin_amdgcn_wmma_f32_16x16x32_f16(false, a0[t][kk], false, bfrag,
                                                             (short)0, acc, false, false);
            }
            #pragma unroll
            for (int v = 0; v < 8; ++v) {
                float val = fmaxf(acc[v] * scale + shift, 0.0f);
                H1[t * 16 + rowhi + v][nn * 16 + col] = (_Float16)val;
            }
        }
    }
    __syncthreads();

    // ---- layer 2 ----
    v16h a1[2][KS2];
    #pragma unroll
    for (int t = 0; t < 2; ++t)
        #pragma unroll
        for (int kk = 0; kk < KS2; ++kk) {
            v16h a;
            #pragma unroll
            for (int e = 0; e < 16; ++e) {
                int K = kk * 32 + kbase + ((e < 8) ? e : e + 8);
                a[e] = H1[t * 16 + arow][K];
            }
            a1[t][kk] = a;
        }

    float* xo = xout + (size_t)gw * C2;
    #pragma unroll 1
    for (int nn = 0; nn < C2 / 16; ++nn) {
        float scale = sc2[nn * 16 + col];
        float shift = sh2[nn * 16 + col];
        float mval = -BIGF;
        #pragma unroll
        for (int t = 0; t < 2; ++t) {
            v8f acc = {};
            #pragma unroll
            for (int kk = 0; kk < KS2; ++kk) {
                v16h bfrag = *(const v16h*)(F2 + ((size_t)(kk * (C2 / 16) + nn)) * 512 + lane * 16);
                acc = __builtin_amdgcn_wmma_f32_16x16x32_f16(false, a1[t][kk], false, bfrag,
                                                             (short)0, acc, false, false);
            }
            #pragma unroll
            for (int v = 0; v < 8; ++v) {
                float val = fmaxf(acc[v] * scale + shift, 0.0f);
                int row = t * 16 + rowhi + v;
                if (!((valid >> row) & 1u)) val = -BIGF;
                mval = fmaxf(mval, val);
            }
        }
        mval = fmaxf(mval, __shfl_xor(mval, 16, 32));
        if (lane < 16) xo[nn * 16 + lane] = mval;
    }
}

// ---------------------------------------------------------------------------
// Global MLP: 16-row tiles of concat[x2, c2] (259 -> pad 288 -> 512 -> 1024),
// max over rows via int-bits atomicMax (post-ReLU values are >= 0).
// ---------------------------------------------------------------------------
__global__ __launch_bounds__(32)
void ga_kernel(const float* __restrict__ x2,   // [B][M2][256]
               const float* __restrict__ c2,   // [B][M2][3]
               const _Float16* __restrict__ F1,
               const float* __restrict__ sc1, const float* __restrict__ sh1,
               const _Float16* __restrict__ F2,
               const float* __restrict__ sc2, const float* __restrict__ sh2,
               float* __restrict__ g)          // [B][1024], zero-initialized
{
    constexpr int CINP = 288, C1 = 512, C2 = 1024;
    constexpr int KS1 = CINP / 32, KS2 = C1 / 32;
    __shared__ _Float16 Araw[16][CINP];
    __shared__ _Float16 H1[16][C1];

    const int lane = threadIdx.x;
    const int rowbase = blockIdx.x * 16;     // flat row = b*M2 + m
    const int b = rowbase / MS2;

    {
        int r = lane & 15;
        const float* xr = x2 + (size_t)(rowbase + r) * 256;
        if (lane < 16) {
            for (int c = 0; c < 128; ++c) Araw[r][c] = (_Float16)xr[c];
        } else {
            for (int c = 128; c < 256; ++c) Araw[r][c] = (_Float16)xr[c];
            const float* cr = c2 + (size_t)(rowbase + r) * 3;
            Araw[r][256] = (_Float16)cr[0];
            Araw[r][257] = (_Float16)cr[1];
            Araw[r][258] = (_Float16)cr[2];
            for (int c = 259; c < CINP; ++c) Araw[r][c] = (_Float16)0.0f;
        }
    }
    __syncthreads();

    const int arow  = lane & 15;
    const int kbase = (lane < 16) ? 0 : 8;
    const int col   = lane & 15;
    const int rowhi = (lane < 16) ? 0 : 8;

    v16h a0[KS1];
    #pragma unroll
    for (int kk = 0; kk < KS1; ++kk) {
        v16h a;
        #pragma unroll
        for (int e = 0; e < 16; ++e) {
            int K = kk * 32 + kbase + ((e < 8) ? e : e + 8);
            a[e] = Araw[arow][K];
        }
        a0[kk] = a;
    }

    #pragma unroll 1
    for (int nn = 0; nn < C1 / 16; ++nn) {
        float scale = sc1[nn * 16 + col];
        float shift = sh1[nn * 16 + col];
        v8f acc = {};
        #pragma unroll
        for (int kk = 0; kk < KS1; ++kk) {
            v16h bfrag = *(const v16h*)(F1 + ((size_t)(kk * (C1 / 16) + nn)) * 512 + lane * 16);
            acc = __builtin_amdgcn_wmma_f32_16x16x32_f16(false, a0[kk], false, bfrag,
                                                         (short)0, acc, false, false);
        }
        #pragma unroll
        for (int v = 0; v < 8; ++v) {
            float val = fmaxf(acc[v] * scale + shift, 0.0f);
            H1[rowhi + v][nn * 16 + col] = (_Float16)val;
        }
    }
    __syncthreads();

    v16h a1[KS2];
    #pragma unroll
    for (int kk = 0; kk < KS2; ++kk) {
        v16h a;
        #pragma unroll
        for (int e = 0; e < 16; ++e) {
            int K = kk * 32 + kbase + ((e < 8) ? e : e + 8);
            a[e] = H1[arow][K];
        }
        a1[kk] = a;
    }

    #pragma unroll 1
    for (int nn = 0; nn < C2 / 16; ++nn) {
        float scale = sc2[nn * 16 + col];
        float shift = sh2[nn * 16 + col];
        v8f acc = {};
        #pragma unroll
        for (int kk = 0; kk < KS2; ++kk) {
            v16h bfrag = *(const v16h*)(F2 + ((size_t)(kk * (C2 / 16) + nn)) * 512 + lane * 16);
            acc = __builtin_amdgcn_wmma_f32_16x16x32_f16(false, a1[kk], false, bfrag,
                                                         (short)0, acc, false, false);
        }
        float mval = 0.0f;
        #pragma unroll
        for (int v = 0; v < 8; ++v)
            mval = fmaxf(mval, fmaxf(acc[v] * scale + shift, 0.0f));
        mval = fmaxf(mval, __shfl_xor(mval, 16, 32));
        if (lane < 16)
            atomicMax((int*)(g + (size_t)b * 1024 + nn * 16 + lane), __float_as_int(mval));
    }
}

// ---------------------------------------------------------------------------
// Head linears (tiny: B=4 rows).
// ---------------------------------------------------------------------------
__global__ void linear_kernel(const float* __restrict__ in, const float* __restrict__ W,
                              const float* __restrict__ bias, float* __restrict__ out,
                              int Cin, int Cout, int relu)
{
    int t = blockIdx.x * blockDim.x + threadIdx.x;
    if (t >= BB * Cout) return;
    int b = t / Cout, o = t % Cout;
    const float* x = in + (size_t)b * Cin;
    float acc = bias[o];
    for (int c = 0; c < Cin; ++c) acc += x[c] * W[(size_t)c * Cout + o];
    if (relu) acc = fmaxf(acc, 0.0f);
    out[(size_t)b * Cout + o] = acc;
}

// ---------------------------------------------------------------------------
extern "C" void kernel_launch(void* const* d_in, const int* in_sizes, int n_in,
                              void* d_out, int out_size, void* d_ws, size_t ws_size,
                              hipStream_t stream)
{
    (void)in_sizes; (void)n_in; (void)out_size; (void)ws_size;
    const float* x   = (const float*)d_in[0];
    const float* pos = (const float*)d_in[1];
    int p = 2;
    // params in setup_inputs() insertion order: sa1(2 blocks), sa2, ga, lin1..3
    const float* sa1W[2]; const float* sa1b[2]; const float* sa1s[2]; const float* sa1t[2];
    const float* sa2W[2]; const float* sa2b[2]; const float* sa2s[2]; const float* sa2t[2];
    const float* gaW[2];  const float* gab[2];  const float* gas[2];  const float* gat[2];
    for (int i = 0; i < 2; ++i) { sa1W[i]=(const float*)d_in[p++]; sa1b[i]=(const float*)d_in[p++];
                                  sa1s[i]=(const float*)d_in[p++]; sa1t[i]=(const float*)d_in[p++]; }
    for (int i = 0; i < 2; ++i) { sa2W[i]=(const float*)d_in[p++]; sa2b[i]=(const float*)d_in[p++];
                                  sa2s[i]=(const float*)d_in[p++]; sa2t[i]=(const float*)d_in[p++]; }
    for (int i = 0; i < 2; ++i) { gaW[i]=(const float*)d_in[p++];  gab[i]=(const float*)d_in[p++];
                                  gas[i]=(const float*)d_in[p++];  gat[i]=(const float*)d_in[p++]; }
    const float* l1W=(const float*)d_in[p++]; const float* l1b=(const float*)d_in[p++];
    const float* l2W=(const float*)d_in[p++]; const float* l2b=(const float*)d_in[p++];
    const float* l3W=(const float*)d_in[p++]; const float* l3b=(const float*)d_in[p++];

    char* wsb = (char*)d_ws;
    size_t off = 0;
    auto alloc = [&](size_t bytes) -> void* {
        void* r = wsb + off;
        off = (off + bytes + 255) & ~(size_t)255;
        return r;
    };

    int*      idx1  = (int*)alloc((size_t)BB * MS1 * 4);
    float*    c1    = (float*)alloc((size_t)BB * MS1 * 3 * 4);
    int*      nidx1 = (int*)alloc((size_t)BB * MS1 * KNB * 4);
    unsigned* vm1   = (unsigned*)alloc((size_t)BB * MS1 * 4);
    float*    x1    = (float*)alloc((size_t)BB * MS1 * 64 * 4);
    int*      idx2  = (int*)alloc((size_t)BB * MS2 * 4);
    float*    c2    = (float*)alloc((size_t)BB * MS2 * 3 * 4);
    int*      nidx2 = (int*)alloc((size_t)BB * MS2 * KNB * 4);
    unsigned* vm2   = (unsigned*)alloc((size_t)BB * MS2 * 4);
    float*    x2    = (float*)alloc((size_t)BB * MS2 * 256 * 4);
    float*    g     = (float*)alloc((size_t)BB * 1024 * 4);
    float*    h1    = (float*)alloc((size_t)BB * 512 * 4);
    float*    h2    = (float*)alloc((size_t)BB * 256 * 4);

    _Float16* fsa1_1 = (_Float16*)alloc((size_t)(32/32)*(32/16)*512*2);
    _Float16* fsa1_2 = (_Float16*)alloc((size_t)(32/32)*(64/16)*512*2);
    _Float16* fsa2_1 = (_Float16*)alloc((size_t)(96/32)*(128/16)*512*2);
    _Float16* fsa2_2 = (_Float16*)alloc((size_t)(128/32)*(256/16)*512*2);
    _Float16* fga_1  = (_Float16*)alloc((size_t)(288/32)*(512/16)*512*2);
    _Float16* fga_2  = (_Float16*)alloc((size_t)(512/32)*(1024/16)*512*2);

    float* sc_sa1[2]; float* sh_sa1[2];
    float* sc_sa2[2]; float* sh_sa2[2];
    float* sc_ga[2];  float* sh_ga[2];
    int cs_sa1[2] = {32, 64}, cs_sa2[2] = {128, 256}, cs_ga[2] = {512, 1024};
    for (int i = 0; i < 2; ++i) { sc_sa1[i]=(float*)alloc(cs_sa1[i]*4); sh_sa1[i]=(float*)alloc(cs_sa1[i]*4); }
    for (int i = 0; i < 2; ++i) { sc_sa2[i]=(float*)alloc(cs_sa2[i]*4); sh_sa2[i]=(float*)alloc(cs_sa2[i]*4); }
    for (int i = 0; i < 2; ++i) { sc_ga[i]=(float*)alloc(cs_ga[i]*4);  sh_ga[i]=(float*)alloc(cs_ga[i]*4); }

    auto repack = [&](const float* W, _Float16* F, int Cin, int Cout, int CinPad) {
        int total = (CinPad / 32) * (Cout / 16) * 512;
        repack_w_kernel<<<(total + 255) / 256, 256, 0, stream>>>(W, F, Cin, Cout, CinPad);
    };
    auto fold = [&](const float* bb, const float* ss, const float* tt, float* sc, float* sh, int C) {
        fold_affine_kernel<<<(C + 255) / 256, 256, 0, stream>>>(bb, ss, tt, sc, sh, C);
    };

    // --- weight prep ---
    repack(sa1W[0], fsa1_1,   6,   32,  32);
    repack(sa1W[1], fsa1_2,  32,   64,  32);
    repack(sa2W[0], fsa2_1,  67,  128,  96);
    repack(sa2W[1], fsa2_2, 128,  256, 128);
    repack(gaW[0],  fga_1,  259,  512, 288);
    repack(gaW[1],  fga_2,  512, 1024, 512);
    for (int i = 0; i < 2; ++i) fold(sa1b[i], sa1s[i], sa1t[i], sc_sa1[i], sh_sa1[i], cs_sa1[i]);
    for (int i = 0; i < 2; ++i) fold(sa2b[i], sa2s[i], sa2t[i], sc_sa2[i], sh_sa2[i], cs_sa2[i]);
    for (int i = 0; i < 2; ++i) fold(gab[i],  gas[i],  gat[i],  sc_ga[i],  sh_ga[i],  cs_ga[i]);

    // --- level 1 ---
    fps_kernel<NPTS, 8><<<BB, NPTS / 8, 0, stream>>>(pos, idx1, MS1);
    gather_centers_kernel<<<(BB * MS1 + 255) / 256, 256, 0, stream>>>(pos, idx1, c1, MS1, NPTS);
    ball_query_kernel<<<(BB * MS1 + 3) / 4, 128, 0, stream>>>(pos, c1, nidx1, vm1, MS1, NPTS, 0.2f * 0.2f);
    pointconv_kernel<32, 32, 64, 3><<<BB * MS1, 32, 0, stream>>>(
        x, pos, c1, nidx1, vm1,
        fsa1_1, sc_sa1[0], sh_sa1[0], fsa1_2, sc_sa1[1], sh_sa1[1],
        x1, MS1, NPTS);

    // --- level 2 ---
    fps_kernel<MS1, 4><<<BB, MS1 / 4, 0, stream>>>(c1, idx2, MS2);
    gather_centers_kernel<<<(BB * MS2 + 255) / 256, 256, 0, stream>>>(c1, idx2, c2, MS2, MS1);
    ball_query_kernel<<<(BB * MS2 + 3) / 4, 128, 0, stream>>>(c1, c2, nidx2, vm2, MS2, MS1, 0.4f * 0.4f);
    pointconv_kernel<96, 128, 256, 64><<<BB * MS2, 32, 0, stream>>>(
        x1, c1, c2, nidx2, vm2,
        fsa2_1, sc_sa2[0], sh_sa2[0], fsa2_2, sc_sa2[1], sh_sa2[1],
        x2, MS2, MS1);

    // --- global MLP + max ---
    zero_kernel<<<(BB * 1024 + 255) / 256, 256, 0, stream>>>(g, BB * 1024);
    ga_kernel<<<BB * MS2 / 16, 32, 0, stream>>>(
        x2, c2, fga_1, sc_ga[0], sh_ga[0], fga_2, sc_ga[1], sh_ga[1], g);

    // --- head ---
    linear_kernel<<<(BB * 512 + 255) / 256, 256, 0, stream>>>(g,  l1W, l1b, h1, 1024, 512, 1);
    linear_kernel<<<(BB * 256 + 255) / 256, 256, 0, stream>>>(h1, l2W, l2b, h2, 512,  256, 1);
    linear_kernel<<<(BB * NCLS + 255) / 256, 256, 0, stream>>>(h2, l3W, l3b, (float*)d_out, 256, NCLS, 0);
}
